// GRUD_34583076667789
// MI455X (gfx1250) — matrix operand order repaired and checked by reading, compiled, not run on hardware
//
#include <hip/hip_runtime.h>
#include <hip/hip_bf16.h>

typedef _Float16 v16h __attribute__((ext_vector_type(16)));
typedef _Float16 v8h  __attribute__((ext_vector_type(8)));
typedef float    v8f  __attribute__((ext_vector_type(8)));
typedef float    v4f  __attribute__((ext_vector_type(4)));

#define NB   128
#define NT   1024
#define NF   512
#define NH   512
#define ROWS 16          // batch rows per workgroup
#define LDH  520         // padded LDS row stride (bank-conflict-free b128)
#define WMAT (NH*NF)     // 262144 elements per weight matrix

__device__ __forceinline__ v16h cat8(v8h lo, v8h hi) {
    return __builtin_shufflevector(lo, hi, 0,1,2,3,4,5,6,7,8,9,10,11,12,13,14,15);
}
__device__ __forceinline__ v8h ld8h(const _Float16* p) { return *(const v8h*)p; }
__device__ __forceinline__ float sigm(float v) { return 1.0f / (1.0f + __expf(-v)); }

// ---------------------------------------------------------------------------
// Convert the four fp32 weight matrices to f16 in workspace (L2-resident).
// ---------------------------------------------------------------------------
__global__ void GRUD_wconv(const float* __restrict__ w0, const float* __restrict__ w1,
                           const float* __restrict__ w2, const float* __restrict__ w3,
                           _Float16* __restrict__ out) {
    int i = blockIdx.x * blockDim.x + threadIdx.x;
    if (i < WMAT) {
        out[i]          = (_Float16)w0[i];
        out[i + WMAT]   = (_Float16)w1[i];
        out[i + 2*WMAT] = (_Float16)w2[i];
        out[i + 3*WMAT] = (_Float16)w3[i];
    }
}

// ---------------------------------------------------------------------------
// Persistent recurrence: one workgroup = 16 batch rows, 16 waves.
// Per step: four [16,512]x[512,512] GEMMs via v_wmma_f32_16x16x32_f16.
// x_{t+1} is double-buffered through registers (loads issued before the GEMM
// phase, committed to LDS in the h-write window) -> 2 barriers/step and no
// x-load latency on the sequential critical path. x_{t+2} is L2-prefetched.
// ---------------------------------------------------------------------------
__launch_bounds__(512, 1)
__global__ void GRUD_recur(const float* __restrict__ x,
                           const float* __restrict__ hx,
                           const _Float16* __restrict__ w16, // [ri, rh, ci, ch]
                           const float* __restrict__ bri, const float* __restrict__ brh,
                           const float* __restrict__ bci, const float* __restrict__ bch,
                           float* __restrict__ out) {
    __shared__ float    h32[ROWS][LDH];
    __shared__ float    x32[ROWS][LDH];
    __shared__ _Float16 h16[ROWS][LDH];
    __shared__ _Float16 x16[ROWS][LDH];

    const int tid  = threadIdx.x;
    const int lane = tid & 31;
    const int wave = tid >> 5;           // 0..15
    const int bg   = blockIdx.x;         // batch group (16 rows)
    const int nl   = lane & 15;          // lane % 16
    const int hi   = lane >> 4;          // lane half selector

    // init h from hx
    for (int i = tid; i < ROWS * NH; i += 512) {
        int m = i >> 9, n = i & 511;
        float v = hx[(size_t)(bg*ROWS + m)*NH + n];
        h32[m][n] = v; h16[m][n] = (_Float16)v;
    }

    const _Float16* Wri = w16;
    const _Float16* Wrh = w16 + WMAT;
    const _Float16* Wci = w16 + 2*WMAT;
    const _Float16* Wch = w16 + 3*WMAT;

    const int n0 = wave * 2;             // this wave owns N-tiles n0, n0+1
    // biases at this lane's output column (C/D layout: n = tile*16 + lane%16)
    float bR[2], bC[2], bH[2];
    #pragma unroll
    for (int tt = 0; tt < 2; ++tt) {
        int n = (n0 + tt)*16 + nl;
        bR[tt] = bri[n] + brh[n];
        bC[tt] = bci[n];
        bH[tt] = bch[n];
    }

    // per-thread x staging slots: pos = i*512 + tid -> (row m, col c4)
    int xm[4], xc[4];
    #pragma unroll
    for (int i = 0; i < 4; ++i) {
        int pos = i*512 + tid;
        xm[i] = pos >> 7;
        xc[i] = (pos & 127) << 2;
    }
    const float* xbase = x + (size_t)(bg*ROWS)*NT*NF;   // rows are NT*NF apart

    // ---- prologue: stage x_0 into LDS ----
    #pragma unroll
    for (int i = 0; i < 4; ++i) {
        v4f v = *(const v4f*)(xbase + (size_t)xm[i]*NT*NF + xc[i]);
        #pragma unroll
        for (int j = 0; j < 4; ++j) {
            x32[xm[i]][xc[i]+j] = v[j];
            x16[xm[i]][xc[i]+j] = (_Float16)v[j];
        }
    }
    __syncthreads();

    for (int t = 0; t < NT; ++t) {
        // ---- issue x_{t+1} loads (consumed after gate math) ----
        v4f xv[4];
        const bool have_next = (t + 1 < NT);
        if (have_next) {
            const float* xn = xbase + (size_t)(t + 1)*NF;
            #pragma unroll
            for (int i = 0; i < 4; ++i)
                xv[i] = *(const v4f*)(xn + (size_t)xm[i]*NT*NF + xc[i]);
        }
        if (t + 2 < NT) {   // pull x_{t+2} toward L2 (global_prefetch_b8)
            const float* xp = xbase + (size_t)(t + 2)*NF;
            #pragma unroll
            for (int i = 0; i < 4; ++i)
                __builtin_prefetch(xp + (size_t)xm[i]*NT*NF + xc[i], 0, 1);
        }

        // ---- four GEMMs: XR=x@Wri^T, HR=h@Wrh^T, XC=x@Wci^T, HC=h@Wch^T ----
        v8f aXR[2] = {}; v8f aHR[2] = {}; v8f aXC[2] = {}; v8f aHC[2] = {};
        const int ka0 = hi * 16;                // A-frag K base (lanes 0-15: K0-15)
        const int kb0 = hi * 16;                // B-frag K base (lanes 0-15: K0-15)
        for (int kc = 0; kc < NF/32; ++kc) {
            int ka = kc*32 + ka0;
            v16h ax = cat8(ld8h(&x16[nl][ka]), ld8h(&x16[nl][ka+8]));
            v16h ah = cat8(ld8h(&h16[nl][ka]), ld8h(&h16[nl][ka+8]));
            int kb = kc*32 + kb0;
            #pragma unroll
            for (int tt = 0; tt < 2; ++tt) {
                size_t row = (size_t)((n0 + tt)*16 + nl) * NF + kb;
                v16h bRi = cat8(ld8h(Wri + row), ld8h(Wri + row + 8));
                v16h bRh = cat8(ld8h(Wrh + row), ld8h(Wrh + row + 8));
                v16h bCi = cat8(ld8h(Wci + row), ld8h(Wci + row + 8));
                v16h bCh = cat8(ld8h(Wch + row), ld8h(Wch + row + 8));
                aXR[tt] = __builtin_amdgcn_wmma_f32_16x16x32_f16(false, ax, false, bRi, (short)0, aXR[tt], false, false);
                aHR[tt] = __builtin_amdgcn_wmma_f32_16x16x32_f16(false, ah, false, bRh, (short)0, aHR[tt], false, false);
                aXC[tt] = __builtin_amdgcn_wmma_f32_16x16x32_f16(false, ax, false, bCi, (short)0, aXC[tt], false, false);
                aHC[tt] = __builtin_amdgcn_wmma_f32_16x16x32_f16(false, ah, false, bCh, (short)0, aHC[tt], false, false);
            }
        }

        // ---- gate math in registers (C/D layout: VGPR v -> M = v + 8*hi) ----
        float hn[2][8];
        #pragma unroll
        for (int tt = 0; tt < 2; ++tt) {
            int n = (n0 + tt)*16 + nl;
            #pragma unroll
            for (int v = 0; v < 8; ++v) {
                int m = v + hi*8;
                float rh = sigm(aXR[tt][v] + aHR[tt][v] + bR[tt]);
                float c  = tanhf(aXC[tt][v] + bC[tt] + rh * (aHC[tt][v] + bH[tt]));
                float hv = h32[m][n];
                float z  = sigm(x32[m][n] + hv);
                hn[tt][v] = z * hv + (1.0f - z) * c;
            }
        }
        __syncthreads();   // all reads of h(t)/x(t) done

        // ---- commit h_{t+1} and x_{t+1} to LDS in the same window ----
        #pragma unroll
        for (int tt = 0; tt < 2; ++tt) {
            int n = (n0 + tt)*16 + nl;
            #pragma unroll
            for (int v = 0; v < 8; ++v) {
                int m = v + hi*8;
                h32[m][n] = hn[tt][v];
                h16[m][n] = (_Float16)hn[tt][v];
            }
        }
        if (have_next) {
            #pragma unroll
            for (int i = 0; i < 4; ++i) {
                #pragma unroll
                for (int j = 0; j < 4; ++j) {
                    x32[xm[i]][xc[i]+j] = xv[i][j];
                    x16[xm[i]][xc[i]+j] = (_Float16)xv[i][j];
                }
            }
        }
        __syncthreads();   // h/x visible before next step's reads
    }

    // ---- emit (h.T, h) ----
    for (int i = tid; i < ROWS * NH; i += 512) {
        int m = i >> 9, n = i & 511;
        float v = h32[m][n];
        int b = bg*ROWS + m;
        out[(size_t)n*NB + b] = v;                       // h.T : [H, B]
        out[(size_t)NB*NH + (size_t)b*NH + n] = v;       // h   : [B, H]
    }
}

extern "C" void kernel_launch(void* const* d_in, const int* in_sizes, int n_in,
                              void* d_out, int out_size, void* d_ws, size_t ws_size,
                              hipStream_t stream) {
    const float* x   = (const float*)d_in[0];
    const float* hx  = (const float*)d_in[1];
    const float* Wri = (const float*)d_in[2];
    const float* bri = (const float*)d_in[3];
    const float* Wrh = (const float*)d_in[4];
    const float* brh = (const float*)d_in[5];
    const float* Wci = (const float*)d_in[6];
    const float* bci = (const float*)d_in[7];
    const float* Wch = (const float*)d_in[8];
    const float* bch = (const float*)d_in[9];

    _Float16* w16 = (_Float16*)d_ws;     // 4 * 512*512 f16 = 2 MB

    GRUD_wconv<<<(WMAT + 255)/256, 256, 0, stream>>>(Wri, Wrh, Wci, Wch, w16);
    GRUD_recur<<<NB/ROWS, 512, 0, stream>>>(x, hx, w16, bri, brh, bci, bch,
                                            (float*)d_out);
}
